// Decoder_33139967656611
// MI455X (gfx1250) — matrix operand order repaired and checked by reading, not compiled, not run on hardware
//
#include <hip/hip_runtime.h>

// ---------------- problem constants ----------------
#define HH     1000
#define H4     4000
#define DICTN  4811
#define TT     44
#define BB     256
// padded dims (K multiples of 32, N multiples of 32 for 2x2 wave blocking)
#define KH     1024          // padded H (K dim)
#define K1     2048          // layer1 input row: [ xt(0..999) pad | h1(1024..2023) pad ]
#define OFF_H1 1024
#define K2     6880          // layer2 row: [ prev(0..4810) | h2 @4832 | h1n @5856 | pad ]
#define OFF2_H2  4832
#define OFF2_H1N 5856
#define H4P    4032          // padded 4H for N (126 x 32)
#define NOUTP  4832          // padded DICT for N (151 x 32)
#define NHP    1024          // padded H for N in xp gemm (32 x 32)
#define MX     (BB*TT)       // 11264

typedef unsigned short u16;
typedef __attribute__((ext_vector_type(16))) __bf16 bf16x16;
typedef __attribute__((ext_vector_type(8)))  float  floatx8;

// ---------------- workspace layout (bytes) ----------------
constexpr size_t alignup(size_t x) { return (x + 255) & ~(size_t)255; }
constexpr size_t OFF_XP  = 0;                                              // [TT][BB][K1] u16
constexpr size_t OFF_A2  = alignup(OFF_XP  + (size_t)TT*BB*K1*2);          // [BB][K2]  u16
constexpr size_t OFF_AO  = alignup(OFF_A2  + (size_t)BB*K2*2);             // [BB][KH]  u16
constexpr size_t OFF_XA  = alignup(OFF_AO  + (size_t)BB*KH*2);             // [MX][KH]  u16
constexpr size_t OFF_W1  = alignup(OFF_XA  + (size_t)MX*KH*2);             // [H4P][K1]  u16
constexpr size_t OFF_W2  = alignup(OFF_W1  + (size_t)H4P*K1*2);            // [H4P][K2]  u16
constexpr size_t OFF_WO  = alignup(OFF_W2  + (size_t)H4P*K2*2);            // [NOUTP][KH] u16
constexpr size_t OFF_WL  = alignup(OFF_WO  + (size_t)NOUTP*KH*2);          // [NHP][KH]  u16
constexpr size_t OFF_B1  = alignup(OFF_WL  + (size_t)NHP*KH*2);            // [H4P] f32
constexpr size_t OFF_B2  = alignup(OFF_B1  + (size_t)H4P*4);               // [H4P] f32
constexpr size_t OFF_OB  = alignup(OFF_B2  + (size_t)H4P*4);               // [NOUTP] f32
constexpr size_t OFF_G1  = alignup(OFF_OB  + (size_t)NOUTP*4);             // [BB][H4P] f32
constexpr size_t OFF_G2  = alignup(OFF_G1  + (size_t)BB*H4P*4);            // [BB][H4P] f32
constexpr size_t OFF_LG  = alignup(OFF_G2  + (size_t)BB*H4P*4);            // [BB][NOUTP] f32
constexpr size_t OFF_C1  = alignup(OFF_LG  + (size_t)BB*NOUTP*4);          // [BB][HH] f32
constexpr size_t OFF_C2  = alignup(OFF_C1  + (size_t)BB*HH*4);             // [BB][HH] f32
constexpr size_t WS_TOTAL = alignup(OFF_C2 + (size_t)BB*HH*4);

// ---------------- helpers ----------------
__device__ __forceinline__ u16 f2bf(float f) {
  union { float f; unsigned u; } x; x.f = f;
  unsigned r = x.u + 0x7FFFu + ((x.u >> 16) & 1u);   // RNE
  return (u16)(r >> 16);
}
__device__ __forceinline__ float sigm(float x) { return 1.f / (1.f + __expf(-x)); }

// WMMA fragment loads per CDNA5 16-bit layouts (wave32).
// A: 16x32, lane m(0..15): VGPR0..3 hold K=2v,2v+1; VGPR4..7 K=16+2v; lanes16-31 shift K by +8.
__device__ __forceinline__ void load_a(const u16* A, int lda, unsigned* u) {
  int lane = threadIdx.x & 31;
  const u16* p = A + (size_t)(lane & 15) * lda + ((lane >> 4) << 3);
#pragma unroll
  for (int v = 0; v < 4; ++v) u[v]     = *(const unsigned*)(p + 2 * v);
#pragma unroll
  for (int v = 0; v < 4; ++v) u[4 + v] = *(const unsigned*)(p + 16 + 2 * v);
}
// B: 32x16 sourced from row-major W[N][K] (B[k][n] = W[n][k]).
// lane n(0..15): VGPR0..3 hold K=2v,2v+1; VGPR4..7 K=16+2v; lanes16-31 shift K by +8.
__device__ __forceinline__ void load_b(const u16* W, int ldw, unsigned* u) {
  int lane = threadIdx.x & 31;
  const u16* p = W + (size_t)(lane & 15) * ldw + ((lane >> 4) << 3);
#pragma unroll
  for (int v = 0; v < 8; ++v) u[v] = *(const unsigned*)(p + 2 * v + ((v >> 2) << 3));
}

union FragBF { bf16x16 v; unsigned u[8]; };

#define WMMA_BF16(A_, B_, C_) \
  __builtin_amdgcn_wmma_f32_16x16x32_bf16(false, (A_), false, (B_), (short)0, (C_), false, false)

// ---------------- weight / input packing ----------------
__global__ void k_pack_w1(const float* Wih1, const float* Whh1, u16* W1) {
  size_t i = (size_t)blockIdx.x * 256 + threadIdx.x;
  if (i >= (size_t)H4P * K1) return;
  int n = (int)(i / K1), k = (int)(i % K1);
  float v = 0.f;
  if (n < H4) {
    if (k < HH) v = Wih1[(size_t)n * 2 * HH + k];
    else if (k >= OFF_H1 && k < OFF_H1 + HH) {
      int j = k - OFF_H1;
      v = Wih1[(size_t)n * 2 * HH + HH + j] + Whh1[(size_t)n * HH + j];
    }
  }
  W1[i] = f2bf(v);
}
__global__ void k_pack_w2(const float* Wih2, const float* Whh2, u16* W2) {
  size_t i = (size_t)blockIdx.x * 256 + threadIdx.x;
  if (i >= (size_t)H4P * K2) return;
  int n = (int)(i / K2), k = (int)(i % K2);
  float v = 0.f;
  const size_t KI = DICTN + 2 * HH;  // 6811
  if (n < H4) {
    if (k < DICTN) v = Wih2[(size_t)n * KI + k];
    else if (k >= OFF2_H2 && k < OFF2_H2 + HH) {
      int j = k - OFF2_H2;
      v = Wih2[(size_t)n * KI + DICTN + j] + Whh2[(size_t)n * HH + j];
    } else if (k >= OFF2_H1N && k < OFF2_H1N + HH) {
      int j = k - OFF2_H1N;
      v = Wih2[(size_t)n * KI + DICTN + HH + j];
    }
  }
  W2[i] = f2bf(v);
}
__global__ void k_pack_wo(const float* outW, u16* WO) {
  size_t i = (size_t)blockIdx.x * 256 + threadIdx.x;
  if (i >= (size_t)NOUTP * KH) return;
  int n = (int)(i / KH), k = (int)(i % KH);
  float v = (n < DICTN && k < HH) ? outW[(size_t)n * HH + k] : 0.f;
  WO[i] = f2bf(v);
}
__global__ void k_pack_wl(const float* layerW, u16* WL) {
  size_t i = (size_t)blockIdx.x * 256 + threadIdx.x;
  if (i >= (size_t)NHP * KH) return;
  int n = (int)(i / KH), k = (int)(i % KH);
  float v = (n < HH && k < HH) ? layerW[(size_t)n * HH + k] : 0.f;
  WL[i] = f2bf(v);
}
__global__ void k_pack_x(const float* x, u16* XA) {
  size_t i = (size_t)blockIdx.x * 256 + threadIdx.x;
  if (i >= (size_t)MX * KH) return;
  int m = (int)(i / KH), k = (int)(i % KH);
  float v = (k < HH) ? x[(size_t)m * HH + k] : 0.f;
  XA[i] = f2bf(v);
}
__global__ void k_bias(const float* a, const float* b, float* o, int n) {
  int i = blockIdx.x * 256 + threadIdx.x;
  if (i < n) o[i] = a[i] + (b ? b[i] : 0.f);
}
__global__ void k_init(const float* h1, const float* h2, const float* c1, const float* c2,
                       u16* xp, u16* a2, float* c1w, float* c2w) {
  int i = blockIdx.x * 256 + threadIdx.x;
  if (i >= BB * HH) return;
  int b = i / HH, j = i % HH;
  xp[(size_t)b * K1 + OFF_H1 + j]  = f2bf(h1[i]);   // step-0 l1 input h-slot
  a2[(size_t)b * K2 + OFF2_H2 + j] = f2bf(h2[i]);   // step-0 l2 input h2-slot
  c1w[i] = c1[i];
  c2w[i] = c2[i];
}

// ---- WMMA GEMM, 2x2 register blocking: each wave computes a 32x32 tile ----
// C[M,N] f32 = A[M,K]bf16 * W[N,K]^T bf16 + bias   (M,N multiples of 32; K mult of 32)
__global__ __launch_bounds__(256) void k_gemm(const u16* A, int lda, const u16* W, int ldw,
                                              const float* bias, float* C, int ldc,
                                              int tilesMB, int tilesNB, int K) {
  int wid = blockIdx.x * (blockDim.x >> 5) + (threadIdx.x >> 5);
  if (wid >= tilesMB * tilesNB) return;        // wave-uniform: EXEC stays all-ones
  int tm = wid % tilesMB, tn = wid / tilesMB;
  const u16* Ab = A + (size_t)tm * 32 * lda;
  const u16* Wb = W + (size_t)tn * 32 * ldw;
  floatx8 acc00 = {}, acc01 = {}, acc10 = {}, acc11 = {};
  for (int k0 = 0; k0 < K; k0 += 32) {
    FragBF a0, a1, b0, b1;
    load_a(Ab + k0, lda, a0.u);
    load_a(Ab + (size_t)16 * lda + k0, lda, a1.u);
    load_b(Wb + k0, ldw, b0.u);
    load_b(Wb + (size_t)16 * ldw + k0, ldw, b1.u);
    acc00 = WMMA_BF16(a0.v, b0.v, acc00);
    acc01 = WMMA_BF16(a0.v, b1.v, acc01);
    acc10 = WMMA_BF16(a1.v, b0.v, acc10);
    acc11 = WMMA_BF16(a1.v, b1.v, acc11);
  }
  int lane = threadIdx.x & 31;
  int nl = lane & 15;
  int m0 = tm * 32 + ((lane >> 4) << 3);       // lanes 16-31 -> rows +8
  int n0 = tn * 32;
  float bv0 = bias[n0 + nl];
  float bv1 = bias[n0 + 16 + nl];
#pragma unroll
  for (int r = 0; r < 8; ++r) {
    C[(size_t)(m0 + r)      * ldc + n0 + nl]      = acc00[r] + bv0;
    C[(size_t)(m0 + r)      * ldc + n0 + 16 + nl] = acc01[r] + bv1;
    C[(size_t)(m0 + 16 + r) * ldc + n0 + nl]      = acc10[r] + bv0;
    C[(size_t)(m0 + 16 + r) * ldc + n0 + 16 + nl] = acc11[r] + bv1;
  }
}

// xp GEMM: 2x2 blocking, bf16 output, remap row m=(b*T+t) -> xpbuf row t*B+b, cols < H
__global__ __launch_bounds__(256) void k_gemm_xp(const u16* A, const u16* W,
                                                 const float* bias, u16* XP) {
  const int tilesMB = MX / 32, tilesNB = NHP / 32;
  int wid = blockIdx.x * (blockDim.x >> 5) + (threadIdx.x >> 5);
  if (wid >= tilesMB * tilesNB) return;
  int tm = wid % tilesMB, tn = wid / tilesMB;
  const u16* Ab = A + (size_t)tm * 32 * KH;
  const u16* Wb = W + (size_t)tn * 32 * KH;
  floatx8 acc00 = {}, acc01 = {}, acc10 = {}, acc11 = {};
  for (int k0 = 0; k0 < KH; k0 += 32) {
    FragBF a0, a1, b0, b1;
    load_a(Ab + k0, KH, a0.u);
    load_a(Ab + (size_t)16 * KH + k0, KH, a1.u);
    load_b(Wb + k0, KH, b0.u);
    load_b(Wb + (size_t)16 * KH + k0, KH, b1.u);
    acc00 = WMMA_BF16(a0.v, b0.v, acc00);
    acc01 = WMMA_BF16(a0.v, b1.v, acc01);
    acc10 = WMMA_BF16(a1.v, b0.v, acc10);
    acc11 = WMMA_BF16(a1.v, b1.v, acc11);
  }
  int lane = threadIdx.x & 31;
  int nl = lane & 15;
  int m0 = tm * 32 + ((lane >> 4) << 3);
  int n0 = tn * 32;
  int na = n0 + nl, nb = n0 + 16 + nl;
  float bva = (na < HH) ? bias[na] : 0.f;      // layer_b has only HH entries
  float bvb = (nb < HH) ? bias[nb] : 0.f;
#pragma unroll
  for (int r = 0; r < 8; ++r) {
    int mA = m0 + r, mB = m0 + 16 + r;
    int bA = mA / TT, tA = mA % TT;
    int bB = mB / TT, tB = mB % TT;
    size_t rowA = ((size_t)tA * BB + bA) * K1;
    size_t rowB = ((size_t)tB * BB + bB) * K1;
    if (na < HH) {
      XP[rowA + na] = f2bf(acc00[r] + bva);
      XP[rowB + na] = f2bf(acc10[r] + bva);
    }
    if (nb < HH) {
      XP[rowA + nb] = f2bf(acc01[r] + bvb);
      XP[rowB + nb] = f2bf(acc11[r] + bvb);
    }
  }
}

// ---------------- LSTM cell (elementwise) ----------------
__global__ void k_cell(const float* gates, float* c, u16* hd1, int ld1, u16* hd2, int ld2) {
  int i = blockIdx.x * 256 + threadIdx.x;
  if (i >= BB * HH) return;
  int b = i / HH, j = i % HH;
  const float* g = gates + (size_t)b * H4P;
  float ig = sigm(g[j]);
  float fg = sigm(g[HH + j]);
  float gg = tanhf(g[2 * HH + j]);
  float og = sigm(g[3 * HH + j]);
  float cn = fg * c[i] + ig * gg;
  c[i] = cn;
  u16 hb = f2bf(og * tanhf(cn));
  hd1[(size_t)b * ld1 + j] = hb;
  if (hd2) hd2[(size_t)b * ld2 + j] = hb;
}

// ---------------- softmax over batch axis (dim 0), write y + bf16 prev ----------------
__global__ void k_softmax(const float* logits, float* out, u16* prev, int t) {
  int d = blockIdx.x * 256 + threadIdx.x;
  if (d >= DICTN) return;
  float mx = -3.0e38f;
  for (int b = 0; b < BB; ++b) mx = fmaxf(mx, logits[(size_t)b * NOUTP + d]);
  float s = 0.f;
  for (int b = 0; b < BB; ++b) s += __expf(logits[(size_t)b * NOUTP + d] - mx);
  float inv = 1.f / s;
  for (int b = 0; b < BB; ++b) {
    float y = __expf(logits[(size_t)b * NOUTP + d] - mx) * inv;
    out[((size_t)b * TT + t) * DICTN + d] = y;
    prev[(size_t)b * K2 + d] = f2bf(y);
  }
}

// ---------------- launcher ----------------
extern "C" void kernel_launch(void* const* d_in, const int* in_sizes, int n_in,
                              void* d_out, int out_size, void* d_ws, size_t ws_size,
                              hipStream_t stream) {
  const float* x      = (const float*)d_in[0];
  const float* h1in   = (const float*)d_in[1];
  const float* h2in   = (const float*)d_in[2];
  const float* c1in   = (const float*)d_in[3];
  const float* c2in   = (const float*)d_in[4];
  const float* layerW = (const float*)d_in[5];
  const float* layerb = (const float*)d_in[6];
  const float* Wih1   = (const float*)d_in[7];
  const float* Whh1   = (const float*)d_in[8];
  const float* bih1   = (const float*)d_in[9];
  const float* bhh1   = (const float*)d_in[10];
  const float* Wih2   = (const float*)d_in[11];
  const float* Whh2   = (const float*)d_in[12];
  const float* bih2   = (const float*)d_in[13];
  const float* bhh2   = (const float*)d_in[14];
  const float* outW   = (const float*)d_in[15];
  const float* outb   = (const float*)d_in[16];
  float* out = (float*)d_out;
  char*  ws  = (char*)d_ws;

  u16*   XP = (u16*)(ws + OFF_XP);
  u16*   A2 = (u16*)(ws + OFF_A2);
  u16*   AO = (u16*)(ws + OFF_AO);
  u16*   XA = (u16*)(ws + OFF_XA);
  u16*   W1 = (u16*)(ws + OFF_W1);
  u16*   W2 = (u16*)(ws + OFF_W2);
  u16*   WO = (u16*)(ws + OFF_WO);
  u16*   WL = (u16*)(ws + OFF_WL);
  float* B1 = (float*)(ws + OFF_B1);
  float* B2 = (float*)(ws + OFF_B2);
  float* OB = (float*)(ws + OFF_OB);
  float* G1 = (float*)(ws + OFF_G1);
  float* G2 = (float*)(ws + OFF_G2);
  float* LG = (float*)(ws + OFF_LG);
  float* C1 = (float*)(ws + OFF_C1);
  float* C2 = (float*)(ws + OFF_C2);

  auto cdiv = [](size_t a, size_t b) { return (unsigned)((a + b - 1) / b); };

  hipMemsetAsync(d_ws, 0, WS_TOTAL, stream);   // zero padding / prev0 / bias pads

  // pack weights + inputs to padded bf16
  k_pack_w1<<<cdiv((size_t)H4P * K1, 256), 256, 0, stream>>>(Wih1, Whh1, W1);
  k_pack_w2<<<cdiv((size_t)H4P * K2, 256), 256, 0, stream>>>(Wih2, Whh2, W2);
  k_pack_wo<<<cdiv((size_t)NOUTP * KH, 256), 256, 0, stream>>>(outW, WO);
  k_pack_wl<<<cdiv((size_t)NHP * KH, 256), 256, 0, stream>>>(layerW, WL);
  k_pack_x <<<cdiv((size_t)MX * KH, 256), 256, 0, stream>>>(x, XA);
  k_bias<<<cdiv(H4, 256), 256, 0, stream>>>(bih1, bhh1, B1, H4);
  k_bias<<<cdiv(H4, 256), 256, 0, stream>>>(bih2, bhh2, B2, H4);
  k_bias<<<cdiv(DICTN, 256), 256, 0, stream>>>(outb, nullptr, OB, DICTN);
  k_init<<<cdiv(BB * HH, 256), 256, 0, stream>>>(h1in, h2in, c1in, c2in, XP, A2, C1, C2);

  // xp = x @ layer_W.T + layer_b  -> bf16 into XP[t][b][0:1000]
  {
    unsigned waves = (MX / 32) * (NHP / 32);
    k_gemm_xp<<<cdiv(waves, 8), 256, 0, stream>>>(XA, WL, layerb, XP);
  }

  const unsigned gblocks = cdiv((size_t)(BB / 32) * (H4P / 32), 8);     // 126 blocks
  const unsigned oblocks = cdiv((size_t)(BB / 32) * (NOUTP / 32), 8);   // 151 blocks

  for (int t = 0; t < TT; ++t) {
    const u16* A1t = XP + (size_t)t * BB * K1;
    // layer1 gates
    k_gemm<<<gblocks, 256, 0, stream>>>(A1t, K1, W1, K1, B1, G1, H4P,
                                        BB / 32, H4P / 32, K1);
    // layer1 cell -> h1 into next step's l1 slot + this step's l2 h1n slot
    if (t < TT - 1) {
      u16* hd1 = XP + (size_t)(t + 1) * BB * K1 + OFF_H1;
      k_cell<<<cdiv(BB * HH, 256), 256, 0, stream>>>(G1, C1, hd1, K1,
                                                     A2 + OFF2_H1N, K2);
    } else {
      k_cell<<<cdiv(BB * HH, 256), 256, 0, stream>>>(G1, C1, A2 + OFF2_H1N, K2,
                                                     nullptr, 0);
    }
    // layer2 gates
    k_gemm<<<gblocks, 256, 0, stream>>>(A2, K2, W2, K2, B2, G2, H4P,
                                        BB / 32, H4P / 32, K2);
    // layer2 cell -> h2 into l2 slot (next step) + AO (for logits)
    k_cell<<<cdiv(BB * HH, 256), 256, 0, stream>>>(G2, C2, A2 + OFF2_H2, K2, AO, KH);
    // logits
    k_gemm<<<oblocks, 256, 0, stream>>>(AO, KH, WO, KH, OB, LG, NOUTP,
                                        BB / 32, NOUTP / 32, KH);
    // softmax over batch, emit y and bf16 prev
    k_softmax<<<cdiv(DICTN, 256), 256, 0, stream>>>(LG, out, A2, t);
  }

  (void)in_sizes; (void)n_in; (void)out_size; (void)ws_size;
}
